// LGCNEncoder_53051436040571
// MI455X (gfx1250) — matrix operand order repaired and checked by reading, not compiled, not run on hardware
//
#include <hip/hip_runtime.h>

#define EMB 64

typedef __attribute__((ext_vector_type(16))) _Float16 v16h;
typedef __attribute__((ext_vector_type(8)))  float    v8f;

__device__ __forceinline__ void atomic_add_f32(float* p, float v) {
  // hardware global_atomic_add_f32 (device scope), no CAS loop
  unsafeAtomicAdd(p, v);
}

// stacked layer 0 = concat(user_emb, item_emb); layers 1..3 = 0
__global__ void lgcn_init_stacked(const float* __restrict__ user_emb,
                                  const float* __restrict__ item_emb,
                                  float* __restrict__ stacked,
                                  int layer_f4, int user_f4) {
  int i = blockIdx.x * blockDim.x + threadIdx.x;   // float4 index over 4 layers
  if (i >= 4 * layer_f4) return;
  float4* out = (float4*)stacked;
  if (i < layer_f4) {
    float4 v = (i < user_f4) ? ((const float4*)user_emb)[i]
                             : ((const float4*)item_emb)[i - user_f4];
    out[i] = v;
  } else {
    out[i] = make_float4(0.f, 0.f, 0.f, 0.f);
  }
}

// one edge per 16 lanes: b128 gather of x[col], 4 hw f32 atomics per lane into out[row]
__global__ void lgcn_spmm_scatter(const int* __restrict__ rows,
                                  const int* __restrict__ cols,
                                  const float* __restrict__ vals,
                                  const float* __restrict__ xin,
                                  float* __restrict__ xout,
                                  int nnz) {
  int t = blockIdx.x * blockDim.x + threadIdx.x;
  int e = t >> 4;
  if (e >= nnz) return;
  int sub = t & 15;
  int r = rows[e];
  int c = cols[e];
  float v = vals[e];
  float4 x = ((const float4*)(xin + c * EMB))[sub];
  float* dst = xout + r * EMB + sub * 4;
  atomic_add_f32(dst + 0, v * x.x);
  atomic_add_f32(dst + 1, v * x.y);
  atomic_add_f32(dst + 2, v * x.z);
  atomic_add_f32(dst + 3, v * x.w);
}

// mean over 4 layers as D = 0.25[I|I] x [S0;S1]  +  0.25[I|I] x [S2;S3]
// two accumulating v_wmma_f32_16x16x32_f16 per 16x16 output tile
__global__ void lgcn_mean_wmma(const float* __restrict__ stacked,
                               float* __restrict__ final_emb,
                               int layer_elems) {
  const int lane  = threadIdx.x;       // 0..31 (wave32)
  const int ntile = threadIdx.y;       // 0..3  (64 cols / 16)
  const int r0 = blockIdx.x * 16;
  const int c0 = ntile * 16;

  // A = 0.25 * [I16 | I16]  (16x32 f16) in wave32 A-operand layout.
  // Per half-index i (compile-time constant -> pure v_cndmask, no scratch):
  //   contiguous K = 16*(lane>=16) + i: nonzero iff (lane & 15) == i
  v16h a;
  #pragma unroll
  for (int i = 0; i < 16; ++i) {
    bool hit = ((lane & 15) == i);
    a[i] = hit ? (_Float16)0.25f : (_Float16)0.0f;
  }

  // B operand (32x16 f16): lane n<16 holds column n, K=0..15 (layer la);
  // lanes 16-31 hold column n-16, K=16..31 (layer lb). Column-major gather.
  const int n = lane & 15;
  const int la01 = (lane < 16) ? 0 : 1;
  const int la23 = (lane < 16) ? 2 : 3;
  const float* p01 = stacked + (long long)la01 * layer_elems + (long long)r0 * EMB + c0 + n;
  const float* p23 = stacked + (long long)la23 * layer_elems + (long long)r0 * EMB + c0 + n;

  v16h b01, b23;
  #pragma unroll
  for (int k = 0; k < 16; ++k) {
    b01[k] = (_Float16)p01[k * EMB];
    b23[k] = (_Float16)p23[k * EMB];
  }

  v8f c = {};
  c = __builtin_amdgcn_wmma_f32_16x16x32_f16(false, a, false, b01, (short)0, c, false, false);
  c = __builtin_amdgcn_wmma_f32_16x16x32_f16(false, a, false, b23, (short)0, c, false, false);

  // D layout: VGPR r holds (M = r + 8*(lane>=16), N = lane&15)
  const int mbase = (lane < 16) ? 0 : 8;
  float* outp = final_emb + (long long)(r0 + mbase) * EMB + c0 + n;
  #pragma unroll
  for (int rr = 0; rr < 8; ++rr) outp[rr * EMB] = c[rr];
}

extern "C" void kernel_launch(void* const* d_in, const int* in_sizes, int n_in,
                              void* d_out, int out_size, void* d_ws, size_t ws_size,
                              hipStream_t stream) {
  const float* user_emb = (const float*)d_in[0];
  const float* item_emb = (const float*)d_in[1];
  const int*   rows     = (const int*)d_in[2];
  const int*   cols     = (const int*)d_in[3];
  const float* vals     = (const float*)d_in[4];

  const int n_users     = in_sizes[0] / EMB;     // 100000
  const int n_items     = in_sizes[1] / EMB;     // 50000
  const int n_nodes     = n_users + n_items;     // 150000
  const int nnz         = in_sizes[2];           // 6000000
  const int layer_elems = n_nodes * EMB;         // 9,600,000

  // d_out = [final (n_nodes*EMB)] ++ [stacked (4*n_nodes*EMB)]
  float* final_emb = (float*)d_out;
  float* stacked   = final_emb + layer_elems;

  {
    int total_f4 = layer_elems;                  // 4 layers * (layer_elems/4) float4s
    int grid = (total_f4 + 255) / 256;
    lgcn_init_stacked<<<grid, 256, 0, stream>>>(user_emb, item_emb, stacked,
                                                layer_elems / 4, n_users * EMB / 4);
  }
  {
    long long threads = (long long)nnz * 16;
    int grid = (int)((threads + 255) / 256);
    for (int l = 0; l < 3; ++l) {
      lgcn_spmm_scatter<<<grid, 256, 0, stream>>>(
          rows, cols, vals,
          stacked + (long long)l * layer_elems,
          stacked + (long long)(l + 1) * layer_elems, nnz);
    }
  }
  {
    dim3 blk(32, 4);                             // 4 waves: one per 16-col tile
    int grid = n_nodes / 16;                     // 9375 row tiles
    lgcn_mean_wmma<<<grid, blk, 0, stream>>>(stacked, final_emb, layer_elems);
  }
}
